// CoupledAttention_51539607937
// MI455X (gfx1250) — matrix operand order repaired and checked by reading, not compiled, not run on hardware
//
#include <hip/hip_runtime.h>
#include <hip/hip_bf16.h>

typedef _Float16 half_t;
typedef __attribute__((ext_vector_type(16))) _Float16 v16h;
typedef __attribute__((ext_vector_type(8)))  float    v8f;

#define NHc 128
#define H2c 64
#define G3c 192     // 3 * H2
#define Sc  512
#define Bc  64

// ---------------------------------------------------------------------------
// WMMA helpers (CDNA5: D = A(16x32 f16) x B(32x16 f16) + C(16x16 f32))
// ---------------------------------------------------------------------------
__device__ __forceinline__ v8f wmma_f16(v16h a, v16h b, v8f c) {
  return __builtin_amdgcn_wmma_f32_16x16x32_f16(false, a, false, b, (short)0, c,
                                                false, false);
}

// A fragment (16x32, MxK) from a row-major f16 matrix.
__device__ __forceinline__ v16h load_a_frag(const half_t* base, int ld, int lane) {
  int row = lane & 15;
  int kb  = (lane >> 4) * 16;
  const half_t* p = base + row * ld + kb;
  v16h f;
#pragma unroll
  for (int i = 0; i < 16; ++i) f[i] = p[i];
  return f;
}

// B fragment (32x16, KxN): B[k][n] = W[n][k], W row-major (outdim, indim).
__device__ __forceinline__ v16h load_b_frag(const half_t* base, int ld, int lane) {
  int row = lane & 15;
  int kb  = (lane >> 4) * 16;
  const half_t* p = base + row * ld + kb;
  v16h f;
#pragma unroll
  for (int i = 0; i < 16; ++i) f[i] = p[i];
  return f;
}

// CDNA5 has V_TANH_F32 (hardware transcendental). Use it for tanh AND sigmoid
// so each gate nonlinearity is a single TRANS op on the recurrent chain.
__device__ __forceinline__ float fast_tanh(float x) {
#if __has_builtin(__builtin_amdgcn_tanhf)
  return __builtin_amdgcn_tanhf(x);
#else
  return tanhf(x);
#endif
}
__device__ __forceinline__ float fast_sigmoid(float x) {
  return 0.5f * fast_tanh(0.5f * x) + 0.5f;
}

// ---------------------------------------------------------------------------
// Kernel 1: w[c][i] = sum_j G[c][0][i][j] * u[j], stored row-major f16 (64,128)
// ---------------------------------------------------------------------------
__global__ void __launch_bounds__(128) prep_w_kernel(
    const float* __restrict__ G_a, const float* __restrict__ G_o,
    const float* __restrict__ u_a, const float* __restrict__ u_o,
    half_t* __restrict__ wF16) {
  int c = blockIdx.x;    // 0..63
  int i = threadIdx.x;   // 0..127
  const float* G;
  const float* u;
  if (c < 32) { G = G_a + (size_t)c * NHc * NHc;        u = u_a; }
  else        { G = G_o + (size_t)(c - 32) * NHc * NHc; u = u_o; }
  float s = 0.f;
#pragma unroll 4
  for (int j = 0; j < NHc; ++j) s += G[i * NHc + j] * u[j];
  wF16[c * NHc + i] = (half_t)s;
}

// ---------------------------------------------------------------------------
// Kernel 2: convert W_ih / W_hh (192x64 f32, row-major) to f16
// ---------------------------------------------------------------------------
__global__ void __launch_bounds__(256) prep_gruw_kernel(
    const float* __restrict__ W_ih, const float* __restrict__ W_hh,
    half_t* __restrict__ WihF16, half_t* __restrict__ WhhF16) {
  int idx = blockIdx.x * blockDim.x + threadIdx.x;
  if (idx < G3c * H2c) {
    WihF16[idx] = (half_t)W_ih[idx];
    WhhF16[idx] = (half_t)W_hh[idx];
  }
}

// ---------------------------------------------------------------------------
// Kernel 3 (fused): beta = tanh(h @ w^T) * mask ; gx = beta @ W_ih^T
// gx scattered to (S,B,192) so each GRU step reads one contiguous 12KB block.
// ---------------------------------------------------------------------------
__global__ void __launch_bounds__(128) beta_gx_kernel(
    const float* __restrict__ h, const unsigned char* __restrict__ mask,
    const half_t* __restrict__ wF16, const half_t* __restrict__ WihF16,
    float* __restrict__ gx) {
  __shared__ half_t hT[16 * NHc];
  __shared__ half_t betaT[16 * 80];   // ld=80 keeps 16B-aligned rows

  const int tid  = threadIdx.x;
  const int wave = tid >> 5;
  const int lane = tid & 31;
  const int m0   = blockIdx.x * 16;

  {
    int r  = tid >> 3;
    int c0 = (tid & 7) * 16;
    const float4* src = (const float4*)(h + (size_t)(m0 + r) * NHc + c0);
    half_t* dst = &hT[r * NHc + c0];
#pragma unroll
    for (int q = 0; q < 4; ++q) {
      float4 f = src[q];
      dst[q * 4 + 0] = (half_t)f.x;
      dst[q * 4 + 1] = (half_t)f.y;
      dst[q * 4 + 2] = (half_t)f.z;
      dst[q * 4 + 3] = (half_t)f.w;
    }
  }
  __syncthreads();

  const int rowadd = (lane >> 4) * 8;
  const int nc     = lane & 15;

  {
    const int n0 = wave * 16;
    v8f acc = {};
#pragma unroll
    for (int kk = 0; kk < NHc; kk += 32) {
      v16h a = load_a_frag(&hT[kk], NHc, lane);
      v16h b = load_b_frag(wF16 + n0 * NHc + kk, NHc, lane);
      acc = wmma_f16(a, b, acc);
    }
#pragma unroll
    for (int vi = 0; vi < 8; ++vi) {
      int m = m0 + vi + rowadd;
      float t = fast_tanh(acc[vi]);
      t *= mask[m] ? 1.0f : 0.0f;
      betaT[(vi + rowadd) * 80 + n0 + nc] = (half_t)t;
    }
  }
  __syncthreads();

#pragma unroll
  for (int q = 0; q < 3; ++q) {
    const int n0 = wave * 16 + q * 64;
    v8f acc = {};
#pragma unroll
    for (int kk = 0; kk < H2c; kk += 32) {
      v16h a = load_a_frag(&betaT[kk], 80, lane);
      v16h b = load_b_frag(WihF16 + n0 * H2c + kk, H2c, lane);
      acc = wmma_f16(a, b, acc);
    }
#pragma unroll
    for (int vi = 0; vi < 8; ++vi) {
      int m  = m0 + vi + rowadd;
      int bb = m >> 9;
      int ss = m & 511;
      gx[((size_t)ss * Bc + bb) * G3c + n0 + nc] = acc[vi];
    }
  }
}

// ---------------------------------------------------------------------------
// Kernel 4: GRU scan. 4 WGs x 16 batch rows, 4 waves each.
// gx[t+1] is DMA'd into an LDS double buffer with global_load_async_to_lds,
// overlapping the step-t gate math + step-(t+1) WMMA (retired via s_wait_asynccnt<=6).
// ---------------------------------------------------------------------------
__device__ __forceinline__ void async_gx_tile(const float* gp, uint32_t lds0,
                                              int wave, int lane) {
  // 16 rows x 192 f32 = 12288 contiguous bytes; 24 b128-instr chunks of 512B.
#pragma unroll
  for (int q = 0; q < 6; ++q) {
    int c = wave * 6 + q;
    unsigned long long ga = (unsigned long long)(uintptr_t)gp +
                            (unsigned long long)c * 512ull +
                            (unsigned long long)lane * 16ull;
    uint32_t la = lds0 + (uint32_t)(c * 512 + lane * 16);
    asm volatile("global_load_async_to_lds_b128 %0, %1, off"
                 :: "v"(la), "v"(ga)
                 : "memory");
  }
}

__global__ void __launch_bounds__(128) gru_kernel(
    const float* __restrict__ gx, const half_t* __restrict__ WhhF16,
    const float* __restrict__ r0, const float* __restrict__ vvec,
    const unsigned char* __restrict__ mask,
    float* __restrict__ outR, float* __restrict__ outY) {
  __shared__ half_t hH[16 * 80];            // f16 state for WMMA A-frags
  __shared__ float  hF[16 * H2c];           // f32 master state
  __shared__ float  ghT[16 * G3c];          // gh = h @ W_hh^T
  __shared__ float  gxb[2][16 * G3c];       // double-buffered gx tiles (2x12KB)
  __shared__ float  vS[H2c];

  const int tid  = threadIdx.x;
  const int wave = tid >> 5;
  const int lane = tid & 31;
  const int b0   = blockIdx.x * 16;

  if (tid < H2c) vS[tid] = vvec[tid];
  for (int idx = tid; idx < 16 * H2c; idx += 128) {
    int j = idx & 63;
    float x = r0[j];
    hF[idx] = x;
    hH[(idx >> 6) * 80 + j] = (half_t)x;
  }

  // Kick off DMA of gx tile for t = 0.
  async_gx_tile(gx + ((size_t)0 * Bc + b0) * G3c,
                (uint32_t)(uintptr_t)&gxb[0][0], wave, lane);
  __syncthreads();

  const int rowadd = (lane >> 4) * 8;
  const int nc     = lane & 15;

  for (int t = 0; t < Sc; ++t) {
    // --- gh = h_prev @ W_hh^T : wave w owns n-tiles {16w, 16w+64, 16w+128}
#pragma unroll
    for (int q = 0; q < 3; ++q) {
      const int n0 = wave * 16 + q * 64;
      v8f acc = {};
#pragma unroll
      for (int kk = 0; kk < H2c; kk += 32) {
        v16h a = load_a_frag(&hH[kk], 80, lane);
        v16h b = load_b_frag(WhhF16 + n0 * H2c + kk, H2c, lane);
        acc = wmma_f16(a, b, acc);
      }
#pragma unroll
      for (int vi = 0; vi < 8; ++vi)
        ghT[(vi + rowadd) * G3c + n0 + nc] = acc[vi];
    }

    // --- issue DMA for gx[t+1]; retire gx[t] (async loads retire in order,
    //     so waiting down to <=6 outstanding means buffer t is resident).
    if (t + 1 < Sc) {
      async_gx_tile(gx + ((size_t)(t + 1) * Bc + b0) * G3c,
                    (uint32_t)(uintptr_t)&gxb[(t + 1) & 1][0], wave, lane);
      asm volatile("s_wait_asynccnt 0x6" ::: "memory");
    } else {
      asm volatile("s_wait_asynccnt 0x0" ::: "memory");
    }
    __syncthreads();

    // --- gates (PyTorch order r, z, n) entirely out of LDS
    const float* gt = &gxb[t & 1][0];
    for (int idx = tid; idx < 16 * H2c; idx += 128) {
      int b  = idx >> 6;
      int j  = idx & 63;
      int br = b0 + b;
      float xr = gt[b * G3c + j];
      float xz = gt[b * G3c + 64 + j];
      float xn = gt[b * G3c + 128 + j];
      float hr = ghT[b * G3c + j];
      float hz = ghT[b * G3c + 64 + j];
      float hn = ghT[b * G3c + 128 + j];
      float rg = fast_sigmoid(xr + hr);
      float zg = fast_sigmoid(xz + hz);
      float ng = fast_tanh(xn + rg * hn);
      float hp = hF[idx];
      float hnew = (1.0f - zg) * ng + zg * hp;
      hF[idx] = hnew;
      hH[b * 80 + j] = (half_t)hnew;
      float mk = mask[(size_t)br * Sc + t] ? 1.0f : 0.0f;
      outR[((size_t)br * Sc + t) * H2c + j] = hnew * mk;
    }
    __syncthreads();

    // --- y[b][t] = r[b][t] . v : 64 threads, 16-wide partials + shfl reduce
    if (tid < 64) {
      int b    = tid >> 2;
      int part = tid & 3;
      float s = 0.f;
#pragma unroll
      for (int j = part * 16; j < part * 16 + 16; ++j)
        s += hF[b * H2c + j] * vS[j];
      s += __shfl_xor(s, 1, 32);
      s += __shfl_xor(s, 2, 32);
      if (part == 0) {
        int br = b0 + b;
        float mk = mask[(size_t)br * Sc + t] ? 1.0f : 0.0f;
        outY[(size_t)br * Sc + t] = s * mk;
      }
    }
    __syncthreads();
  }
}

// ---------------------------------------------------------------------------
// Launch
// ---------------------------------------------------------------------------
extern "C" void kernel_launch(void* const* d_in, const int* in_sizes, int n_in,
                              void* d_out, int out_size, void* d_ws, size_t ws_size,
                              hipStream_t stream) {
  const float*         h    = (const float*)d_in[0];
  const float*         u_a  = (const float*)d_in[1];
  const float*         u_o  = (const float*)d_in[2];
  const unsigned char* mask = (const unsigned char*)d_in[3];
  const float*         G_a  = (const float*)d_in[4];
  const float*         G_o  = (const float*)d_in[5];
  const float*         r0   = (const float*)d_in[6];
  const float*         v    = (const float*)d_in[7];
  const float*         W_ih = (const float*)d_in[8];
  const float*         W_hh = (const float*)d_in[9];

  char* ws = (char*)d_ws;
  half_t* wF16   = (half_t*)(ws + 0);          // 16 KB
  half_t* WihF16 = (half_t*)(ws + 16384);      // 24 KB
  half_t* WhhF16 = (half_t*)(ws + 40960);      // 24 KB
  float*  gx     = (float*)(ws + 65536);       // 512*64*192*4 = 24 MB

  float* outR = (float*)d_out;                     // (B,S,64)
  float* outY = outR + (size_t)Bc * Sc * H2c;      // (B,S)

  prep_w_kernel<<<64, 128, 0, stream>>>(G_a, G_o, u_a, u_o, wF16);
  prep_gruw_kernel<<<48, 256, 0, stream>>>(W_ih, W_hh, WihF16, WhhF16);
  beta_gx_kernel<<<(Bc * Sc) / 16, 128, 0, stream>>>(h, mask, wF16, WihF16, gx);
  gru_kernel<<<Bc / 16, 128, 0, stream>>>(gx, WhhF16, r0, v, mask, outR, outY);
}
